// WindowMultiHeadAttention_8821862826386
// MI455X (gfx1250) — compile-verified, matrix operand order checked by reading, not run on hardware
//
#include <hip/hip_runtime.h>
#include <hip/hip_bf16.h>
#include <math.h>

// ---------------- types ----------------
typedef __attribute__((ext_vector_type(16))) __bf16 v16bf;
typedef __attribute__((ext_vector_type(8)))  __bf16 v8bf;
typedef __attribute__((ext_vector_type(8)))  float  v8f;

// ---------------- problem constants ----------------
#define LW        144              // window area (12*12)
#define C_DIM     256
#define NHEADS    8
#define DH        32
#define NWIN      64
#define BW        1024
#define M_TOT     (BW * LW)        // 147456
#define K_DIM     256
#define N_QKV     768
#define META_H    384
#define LOGIT_MAX 4.6051701859880914f   // log(1/0.01)

// LDS padded strides (elements). chosen so 16B/lane fragment reads hit
// distinct bank groups: stride/4B-dwords * lane mod 64 all-distinct.
#define KSTRIDE   40               // k-tile rows (80B  = 20 dwords/row)
#define VSTRIDE   168              // v^T rows    (336B = 84 dwords/row)
#define PSTRIDE   168              // P rows      (336B)

// =====================================================================
// WMMA helper: D = A(16x32 bf16) * B(32x16 bf16) + C(16x16 f32)
// =====================================================================
static __device__ __forceinline__ v8f wmma_bf16(v16bf a, v16bf b, v8f c) {
  return __builtin_amdgcn_wmma_f32_16x16x32_bf16(
      /*neg_a=*/false, a, /*neg_b=*/false, b,
      /*c_mod=*/(short)0, c, /*reuse_a=*/false, /*reuse_b=*/false);
}

// Load one A/B fragment for 16x16x32 bf16 WMMA from a row-major matrix
// whose rows are contiguous in K. `rowp` points at (row, k0) for this lane
// (row = base + lane%16). Per ISA layout:
//   lanes 0-15 : elems 0..7 = K 0..7 , elems 8..15 = K 16..23
//   lanes 16-31: elems 0..7 = K 8..15, elems 8..15 = K 24..31
static __device__ __forceinline__ v16bf ldfrag(const __bf16* rowp, int lane) {
  const int kb = (lane < 16) ? 0 : 8;
  v8bf lo = *reinterpret_cast<const v8bf*>(rowp + kb);
  v8bf hi = *reinterpret_cast<const v8bf*>(rowp + kb + 16);
  v16bf r;
#pragma unroll
  for (int i = 0; i < 8; ++i) { r[i] = lo[i]; r[i + 8] = hi[i]; }
  return r;
}

// cross-lane reductions over 16-lane groups (matches C/D tile layout:
// a matrix row lives in one group of 16 lanes at a fixed element index)
static __device__ __forceinline__ float rmax16(float v) {
#pragma unroll
  for (int m = 8; m >= 1; m >>= 1) v = fmaxf(v, __shfl_xor(v, m, 16));
  return v;
}
static __device__ __forceinline__ float rsum16(float v) {
#pragma unroll
  for (int m = 8; m >= 1; m >>= 1) v += __shfl_xor(v, m, 16);
  return v;
}

// =====================================================================
// Kernel 0: fp32 -> bf16 conversion (grid-stride)
// =====================================================================
__global__ void f32_to_bf16_kernel(const float* __restrict__ in,
                                   __bf16* __restrict__ out, size_t n) {
  size_t i = (size_t)blockIdx.x * blockDim.x + threadIdx.x;
  size_t stride = (size_t)gridDim.x * blockDim.x;
  for (; i < n; i += stride) out[i] = (__bf16)in[i];
}

// =====================================================================
// Kernel 1: continuous relative-position bias via meta MLP
//   rel_coords_log(p,2) -> relu(fc1) -> fc2  => rel_bias[H][LW][LW] (f32)
// =====================================================================
__global__ void relbias_kernel(const float* __restrict__ fc1_w,  // (384,2)
                               const float* __restrict__ fc1_b,  // (384)
                               const float* __restrict__ fc2_w,  // (8,384)
                               const float* __restrict__ fc2_b,  // (8)
                               float* __restrict__ rel_bias) {   // (8,144,144)
  int p = blockIdx.x * blockDim.x + threadIdx.x;
  if (p >= LW * LW) return;
  int i = p / LW, j = p % LW;
  float d0 = (float)(i / 12 - j / 12);
  float d1 = (float)(i % 12 - j % 12);
  float c0 = copysignf(log1pf(fabsf(d0)), d0);   // sign(d)*log1p(|d|)
  float c1 = copysignf(log1pf(fabsf(d1)), d1);
  float acc[NHEADS];
#pragma unroll
  for (int t = 0; t < NHEADS; ++t) acc[t] = fc2_b[t];
  for (int m = 0; m < META_H; ++m) {
    float h = fmaf(c0, fc1_w[2 * m], fmaf(c1, fc1_w[2 * m + 1], fc1_b[m]));
    h = fmaxf(h, 0.0f);
#pragma unroll
    for (int t = 0; t < NHEADS; ++t) acc[t] = fmaf(fc2_w[t * META_H + m], h, acc[t]);
  }
#pragma unroll
  for (int t = 0; t < NHEADS; ++t) rel_bias[t * LW * LW + p] = acc[t];
}

// =====================================================================
// Kernel 2: bf16 WMMA GEMM  C[M,N] = A[M,K] * W[N,K]^T + bias[N]
// Block: 256 threads = 8 waves (4M x 2N), tile 128(M) x 128(N), K-step 32.
// Each wave: 32x64 = 2x4 accumulator tiles, 8 WMMAs per k-step.
// Software-pipelined: double-buffered LDS, next tile staged in registers
// while WMMAs run, K+2 tile prefetched into cache (global_prefetch_b8).
// =====================================================================
template <typename OutT>
__global__ void __launch_bounds__(256)
gemm_bf16_kernel(const __bf16* __restrict__ A,   // (M,K) bf16
                 const __bf16* __restrict__ W,   // (N,K) bf16
                 const float*  __restrict__ bias,// (N)
                 OutT* __restrict__ Cout,        // (M,N)
                 int K, int N) {
  __shared__ __bf16 As[2][128 * KSTRIDE];   // 2 x 10 KB
  __shared__ __bf16 Bs[2][128 * KSTRIDE];   // 2 x 10 KB

  const int tid  = threadIdx.x;
  const int lane = tid & 31;
  const int wv   = tid >> 5;
  const int lr   = lane & 15;
  const int hi   = lane >> 4;
  const int mo   = (wv >> 1) * 32;   // wave M origin in tile
  const int no   = (wv & 1) * 64;    // wave N origin in tile
  const size_t bm = (size_t)blockIdx.y * 128;
  const size_t bn = (size_t)blockIdx.x * 128;

  // per-thread staging chunks: 512 16B chunks per 128x32 tile, 2 per thread
  const int r0  = tid >> 2;          // rows r0 and r0+64
  const int cc8 = (tid & 3) * 8;     // k-offset within tile (elements)
  const __bf16* aptr0 = A + (bm + r0) * (size_t)K + cc8;
  const __bf16* aptr1 = A + (bm + r0 + 64) * (size_t)K + cc8;
  const __bf16* bptr0 = W + (bn + r0) * (size_t)K + cc8;
  const __bf16* bptr1 = W + (bn + r0 + 64) * (size_t)K + cc8;
  const int soff0 = r0 * KSTRIDE + cc8;
  const int soff1 = (r0 + 64) * KSTRIDE + cc8;

  v8f acc[2][4];
#pragma unroll
  for (int mt = 0; mt < 2; ++mt)
#pragma unroll
    for (int nt = 0; nt < 4; ++nt)
#pragma unroll
      for (int e = 0; e < 8; ++e) acc[mt][nt][e] = 0.0f;

  v8bf ra0, ra1, rb0, rb1;
  // prologue: stage tile 0
  ra0 = *reinterpret_cast<const v8bf*>(aptr0);
  ra1 = *reinterpret_cast<const v8bf*>(aptr1);
  rb0 = *reinterpret_cast<const v8bf*>(bptr0);
  rb1 = *reinterpret_cast<const v8bf*>(bptr1);
  *reinterpret_cast<v8bf*>(&As[0][soff0]) = ra0;
  *reinterpret_cast<v8bf*>(&As[0][soff1]) = ra1;
  *reinterpret_cast<v8bf*>(&Bs[0][soff0]) = rb0;
  *reinterpret_cast<v8bf*>(&Bs[0][soff1]) = rb1;
  __syncthreads();

  const int KS = K >> 5;             // number of 32-wide K steps
  for (int s = 0; s < KS; ++s) {
    const int cur = s & 1;
    // stage next tile into registers (overlaps with WMMAs below)
    if (s + 1 < KS) {
      const int k1 = (s + 1) << 5;
      ra0 = *reinterpret_cast<const v8bf*>(aptr0 + k1);
      ra1 = *reinterpret_cast<const v8bf*>(aptr1 + k1);
      rb0 = *reinterpret_cast<const v8bf*>(bptr0 + k1);
      rb1 = *reinterpret_cast<const v8bf*>(bptr1 + k1);
      if (s + 2 < KS) {              // warm cache for K+2 tile
        const int k2 = (s + 2) << 5;
        __builtin_prefetch(aptr0 + k2, 0, 3);
        __builtin_prefetch(aptr1 + k2, 0, 3);
        __builtin_prefetch(bptr0 + k2, 0, 3);
        __builtin_prefetch(bptr1 + k2, 0, 3);
      }
    }

    v16bf af[2], bf[4];
#pragma unroll
    for (int mt = 0; mt < 2; ++mt)
      af[mt] = ldfrag(&As[cur][(mo + mt * 16 + lr) * KSTRIDE], lane);
#pragma unroll
    for (int nt = 0; nt < 4; ++nt)
      bf[nt] = ldfrag(&Bs[cur][(no + nt * 16 + lr) * KSTRIDE], lane);
#pragma unroll
    for (int mt = 0; mt < 2; ++mt)
#pragma unroll
      for (int nt = 0; nt < 4; ++nt)
        acc[mt][nt] = wmma_bf16(af[mt], bf[nt], acc[mt][nt]);

    // commit next tile to the other LDS buffer, single barrier per step
    if (s + 1 < KS) {
      const int nxt = cur ^ 1;
      *reinterpret_cast<v8bf*>(&As[nxt][soff0]) = ra0;
      *reinterpret_cast<v8bf*>(&As[nxt][soff1]) = ra1;
      *reinterpret_cast<v8bf*>(&Bs[nxt][soff0]) = rb0;
      *reinterpret_cast<v8bf*>(&Bs[nxt][soff1]) = rb1;
      __syncthreads();
    }
  }

  // ---- epilogue: +bias, store (bf16 for qkv path, f32 for proj path) ----
#pragma unroll
  for (int mt = 0; mt < 2; ++mt)
#pragma unroll
    for (int nt = 0; nt < 4; ++nt) {
      int col = (int)bn + no + nt * 16 + lr;
      float bv = bias[col];
#pragma unroll
      for (int j = 0; j < 8; ++j) {
        size_t row = bm + mo + mt * 16 + j + 8 * hi;
        Cout[row * (size_t)N + col] = (OutT)(acc[mt][nt][j] + bv);
      }
    }
}

// =====================================================================
// Kernel 3: windowed cosine attention, one block per (window b, head h)
// 288 threads = 9 waves; wave w owns row strip [16w, 16w+16).
//   S = (q/|q|)(k/|k|)^T * exp(min(ls,LOGIT_MAX)) + rel_bias[h] + mask[b%64]
//   P = softmax(S);  O = P V
// =====================================================================
__global__ void __launch_bounds__(288)
attn_kernel(const __bf16* __restrict__ qkv,        // (M_TOT, 768) bf16
            const float*  __restrict__ mask,       // (64,144,144)
            const float*  __restrict__ rel_bias,   // (8,144,144)
            const float*  __restrict__ logit_scale,// (8)
            __bf16* __restrict__ obuf) {           // (M_TOT, 256) bf16
  extern __shared__ char smem[];
  __bf16* kb = reinterpret_cast<__bf16*>(smem);          // 144 x KSTRIDE
  __bf16* vt = kb + LW * KSTRIDE;                        // 32  x VSTRIDE (v^T, K-padded)
  __bf16* pb = vt + 32 * VSTRIDE;                        // 144 x PSTRIDE (P, K-padded)
  float*  rq = reinterpret_cast<float*>(pb + LW * PSTRIDE); // 144
  float*  rk = rq + LW;                                     // 144

  const int b    = blockIdx.x >> 3;
  const int h    = blockIdx.x & 7;
  const int tid  = threadIdx.x;
  const int lane = tid & 31;
  const int wv   = tid >> 5;          // 0..8
  const int lr   = lane & 15;
  const int hi   = lane >> 4;

  const __bf16* qbase = qkv + (size_t)b * LW * N_QKV + h * DH;       // s=0
  const __bf16* kbase = qbase + C_DIM;                               // s=1
  const __bf16* vbase = qbase + 2 * C_DIM;                           // s=2
  const float* biasb = rel_bias + (size_t)h * LW * LW;
  const float* maskb = mask + (size_t)(b & (NWIN - 1)) * LW * LW;

  // ---------------- phase 1: cooperative staging ----------------
  // k rows -> LDS (144 rows x 32, 576 16B chunks / 288 threads)
#pragma unroll
  for (int i = 0; i < 2; ++i) {
    int c = tid + i * 288, r = c >> 2, cc = c & 3;
    *reinterpret_cast<v8bf*>(&kb[r * KSTRIDE + cc * 8]) =
        *reinterpret_cast<const v8bf*>(kbase + (size_t)r * N_QKV + cc * 8);
  }
  // v -> LDS transposed (vt[d][l]); 4608 elems / 288 threads
#pragma unroll
  for (int i = 0; i < 16; ++i) {
    int e = tid + i * 288, r = e >> 5, d = e & 31;
    vt[d * VSTRIDE + r] = vbase[(size_t)r * N_QKV + d];
  }
  // zero K-pad of v^T (cols 144..167) and of P (cols 144..167)
#pragma unroll
  for (int i = 0; i < 3; ++i) {
    int e = tid + i * 288;
    if (e < 32 * 24) vt[(e / 24) * VSTRIDE + 144 + e % 24] = (__bf16)0.0f;
  }
#pragma unroll
  for (int i = 0; i < 12; ++i) {
    int e = tid + i * 288;
    pb[(e / 24) * PSTRIDE + 144 + e % 24] = (__bf16)0.0f;
  }
  // per-row inverse L2 norms of q and k (cosine attention)
  if (tid < LW) {
    const __bf16* rp = qbase + (size_t)tid * N_QKV;
    float s = 0.0f;
#pragma unroll
    for (int d = 0; d < DH; ++d) { float f = (float)rp[d]; s = fmaf(f, f, s); }
    rq[tid] = 1.0f / fmaxf(sqrtf(s), 1e-12f);
  } else {
    int t = tid - LW;
    const __bf16* rp = kbase + (size_t)t * N_QKV;
    float s = 0.0f;
#pragma unroll
    for (int d = 0; d < DH; ++d) { float f = (float)rp[d]; s = fmaf(f, f, s); }
    rk[t] = 1.0f / fmaxf(sqrtf(s), 1e-12f);
  }
  __syncthreads();

  // ---------------- phase 2: S = q k^T (9 WMMAs per wave) ----------------
  const int m0 = wv * 16;
  // warm cache for the bias/mask rows this wave consumes in the epilogue
  __builtin_prefetch(biasb + (m0 + lr) * LW + hi * 72, 0, 3);
  __builtin_prefetch(maskb + (m0 + lr) * LW + hi * 72, 0, 3);

  v16bf aq = ldfrag(qbase + (size_t)(m0 + lr) * N_QKV, lane);  // q rows direct from global
  v8f acc[9];
#pragma unroll
  for (int jt = 0; jt < 9; ++jt) {
#pragma unroll
    for (int e = 0; e < 8; ++e) acc[jt][e] = 0.0f;
    v16bf bk = ldfrag(&kb[(jt * 16 + lr) * KSTRIDE], lane);
    acc[jt] = wmma_bf16(aq, bk, acc[jt]);
  }

  // ---------------- phase 3: scale + bias + mask + softmax ----------------
  const float scale = __expf(fminf(logit_scale[h], LOGIT_MAX));

#pragma unroll
  for (int jt = 0; jt < 9; ++jt)
#pragma unroll
    for (int j = 0; j < 8; ++j) {
      int row = m0 + j + 8 * hi;
      int col = jt * 16 + lr;
      acc[jt][j] = acc[jt][j] * (rq[row] * rk[col] * scale)
                 + biasb[row * LW + col] + maskb[row * LW + col];
    }

  float rinv[8];
#pragma unroll
  for (int j = 0; j < 8; ++j) {
    float m = -INFINITY;
#pragma unroll
    for (int jt = 0; jt < 9; ++jt) m = fmaxf(m, acc[jt][j]);
    m = rmax16(m);
    float s = 0.0f;
#pragma unroll
    for (int jt = 0; jt < 9; ++jt) {
      float p = __expf(acc[jt][j] - m);
      acc[jt][j] = p;
      s += p;
    }
    s = rsum16(s);
    rinv[j] = 1.0f / s;
  }
  // write P strip (bf16) to this wave's LDS rows; same-wave LDS is in-order
#pragma unroll
  for (int jt = 0; jt < 9; ++jt)
#pragma unroll
    for (int j = 0; j < 8; ++j)
      pb[(m0 + j + 8 * hi) * PSTRIDE + jt * 16 + lr] = (__bf16)(acc[jt][j] * rinv[j]);

  // ---------------- phase 4: O = P V (5 k-steps x 2 WMMAs) ----------------
  v8f o0, o1;
#pragma unroll
  for (int e = 0; e < 8; ++e) { o0[e] = 0.0f; o1[e] = 0.0f; }
#pragma unroll
  for (int ks = 0; ks < 5; ++ks) {   // K = 160 (144 + zero pad)
    v16bf ap  = ldfrag(&pb[(m0 + lr) * PSTRIDE + ks * 32], lane);
    v16bf bv0 = ldfrag(&vt[lr * VSTRIDE + ks * 32], lane);
    v16bf bv1 = ldfrag(&vt[(16 + lr) * VSTRIDE + ks * 32], lane);
    o0 = wmma_bf16(ap, bv0, o0);
    o1 = wmma_bf16(ap, bv1, o1);
  }

  // ---------------- phase 5: store O as (m, h*32+d) bf16 ----------------
  __bf16* ob = obuf + (size_t)b * LW * C_DIM + h * DH;
#pragma unroll
  for (int j = 0; j < 8; ++j) {
    int row = m0 + j + 8 * hi;
    ob[(size_t)row * C_DIM + lr]      = (__bf16)o0[j];
    ob[(size_t)row * C_DIM + 16 + lr] = (__bf16)o1[j];
  }
}

// =====================================================================
// host-side launcher
// =====================================================================
extern "C" void kernel_launch(void* const* d_in, const int* in_sizes, int n_in,
                              void* d_out, int out_size, void* d_ws, size_t ws_size,
                              hipStream_t stream) {
  const float* x           = (const float*)d_in[0];
  const float* mask        = (const float*)d_in[1];
  const float* qkv_w       = (const float*)d_in[2];
  const float* qkv_b       = (const float*)d_in[3];
  const float* proj_w      = (const float*)d_in[4];
  const float* proj_b      = (const float*)d_in[5];
  const float* fc1_w       = (const float*)d_in[6];
  const float* fc1_b       = (const float*)d_in[7];
  const float* fc2_w       = (const float*)d_in[8];
  const float* fc2_b       = (const float*)d_in[9];
  const float* logit_scale = (const float*)d_in[10];
  float* out = (float*)d_out;

  // workspace carve-up (256B aligned)
  char* ws = (char*)d_ws;
  size_t off = 0;
  auto take = [&](size_t bytes) { char* p = ws + off; off = (off + bytes + 255) & ~(size_t)255; return p; };
  __bf16* xb     = (__bf16*)take((size_t)M_TOT * K_DIM * 2);     //  75.5 MB
  __bf16* wqkvb  = (__bf16*)take((size_t)N_QKV * K_DIM * 2);     //   0.4 MB
  __bf16* wprojb = (__bf16*)take((size_t)C_DIM * K_DIM * 2);     //   0.1 MB
  __bf16* qkvb   = (__bf16*)take((size_t)M_TOT * N_QKV * 2);     // 226.5 MB
  __bf16* obuf   = (__bf16*)take((size_t)M_TOT * C_DIM * 2);     //  75.5 MB
  float*  relb   = (float*)take((size_t)NHEADS * LW * LW * 4);   //   0.7 MB

  // 0) fp32 -> bf16
  f32_to_bf16_kernel<<<2048, 256, 0, stream>>>(x, xb, (size_t)M_TOT * K_DIM);
  f32_to_bf16_kernel<<<192, 256, 0, stream>>>(qkv_w, wqkvb, (size_t)N_QKV * K_DIM);
  f32_to_bf16_kernel<<<64, 256, 0, stream>>>(proj_w, wprojb, (size_t)C_DIM * K_DIM);

  // 1) relative position bias (meta MLP)
  relbias_kernel<<<(LW * LW + 255) / 256, 256, 0, stream>>>(fc1_w, fc1_b, fc2_w, fc2_b, relb);

  // 2) QKV projection: (147456,256) x (256,768) -> bf16
  gemm_bf16_kernel<__bf16><<<dim3(N_QKV / 128, M_TOT / 128), 256, 0, stream>>>(
      xb, wqkvb, qkv_b, qkvb, K_DIM, N_QKV);

  // 3) attention: one block per (window, head)
  size_t attn_lds = (size_t)(LW * KSTRIDE + 32 * VSTRIDE + LW * PSTRIDE) * 2 + 2 * LW * 4;
  attn_kernel<<<BW * NHEADS, 288, attn_lds, stream>>>(qkvb, mask, relb, logit_scale, obuf);

  // 4) output projection: (147456,256) x (256,256) -> f32 out
  gemm_bf16_kernel<float><<<dim3(C_DIM / 128, M_TOT / 128), 256, 0, stream>>>(
      obuf, wprojb, proj_b, out, K_DIM, C_DIM);
}
